// HSIViT_86921548136826
// MI455X (gfx1250) — compile-verified, matrix operand order checked by reading, not run on hardware
//
#include <hip/hip_runtime.h>
#include <hip/hip_bf16.h>

#define DEPTH 12
#define CDIM  768
#define NHEAD 12
#define TB    8
#define HD    64
#define NTOK  512
#define BATCH 8
#define MROWS (BATCH * NTOK)   /* 4096 */
#define FF    (4 * CDIM)       /* 3072 */

typedef __attribute__((ext_vector_type(16))) _Float16 v16h;
typedef __attribute__((ext_vector_type(8)))  _Float16 v8h;
typedef __attribute__((ext_vector_type(8)))  float    v8f;

// ---------------------------------------------------------------------------
// WMMA helpers (CDNA5 16x16x32 f16, fp32 accumulate)
// A-fragment layout (ISA 7.12.2, 16-bit A 16x32): lane l<16 -> row l, K
// chunks {0..7, 16..23}; lane>=16 -> row l-16, K chunks {8..15, 24..31}.
// B = W row-major (rows = output features) uses the identical pattern.
// ---------------------------------------------------------------------------
__device__ __forceinline__ v16h ld_frag(const _Float16* tile, int stride, int lane) {
  const int l  = lane & 15;
  const int ko = (lane & 16) >> 1;           // 0 or 8
  const _Float16* p = tile + l * stride + ko;
  v8h a0 = *(const v8h*)(p);
  v8h a1 = *(const v8h*)(p + 16);
  return __builtin_shufflevector(a0, a1, 0,1,2,3,4,5,6,7,8,9,10,11,12,13,14,15);
}

__device__ __forceinline__ v8f wmma_f16(v16h a, v16h b, v8f c) {
  return __builtin_amdgcn_wmma_f32_16x16x32_f16(false, a, false, b, (short)0, c,
                                                false, false);
}

__device__ __forceinline__ v8h cvt8(float4 f0, float4 f1) {
  v8h h;
  h[0] = (_Float16)f0.x; h[1] = (_Float16)f0.y; h[2] = (_Float16)f0.z; h[3] = (_Float16)f0.w;
  h[4] = (_Float16)f1.x; h[5] = (_Float16)f1.y; h[6] = (_Float16)f1.z; h[7] = (_Float16)f1.w;
  return h;
}

__device__ __forceinline__ void cvt_store8(_Float16* dst, float4 f0, float4 f1) {
  *(v8h*)dst = cvt8(f0, f1);
}

// ---------------------------------------------------------------------------
// Generic fused GEMM: out[M,N] = act(A[M,K] @ W[N,K]^T + bias[N]) (+ res)
// ACT: 0 = none, 1 = exact GELU.  RES: 1 = add residual (res may alias out).
// Block 256 thr (8 waves, 2x4), block tile 128x256, wave tile 64x64,
// K step 32, double-buffered LDS (one barrier/step), global prefetch of the
// next k-tile (converted to f16 in regs) overlapped with the WMMA burst.
// ---------------------------------------------------------------------------
template<int ACT, int RES>
__global__ __launch_bounds__(256) void gemm_kernel(
    const float* __restrict__ A, const float* __restrict__ W,
    const float* __restrict__ bias, const float* __restrict__ res,
    float* __restrict__ out, int M, int N, int K)
{
  // rows 0..127: A tile, rows 128..383: W tile; 80B pitch -> conflict-free b128
  __shared__ _Float16 Sh[2][384 * 40];

  const int tid = threadIdx.x, lane = tid & 31, wid = tid >> 5;
  const int wm = wid >> 2, wn = wid & 3;
  const int bm = blockIdx.y * 128, bn = blockIdx.x * 256;
  const int lr = tid >> 1, ch = (tid & 1) * 16;

  const float* gA  = A + (size_t)(bm + lr) * K + ch;
  const float* gB0 = W + (size_t)(bn + lr) * K + ch;
  const float* gB1 = W + (size_t)(bn + 128 + lr) * K + ch;

  v8f acc[4][4] = {};
  v8h pre[6];

  { // prefetch k-tile 0
    const float4* a  = (const float4*)gA;
    const float4* b0 = (const float4*)gB0;
    const float4* b1 = (const float4*)gB1;
    pre[0] = cvt8(a[0],  a[1]);  pre[1] = cvt8(a[2],  a[3]);
    pre[2] = cvt8(b0[0], b0[1]); pre[3] = cvt8(b0[2], b0[3]);
    pre[4] = cvt8(b1[0], b1[1]); pre[5] = cvt8(b1[2], b1[3]);
  }
  { // stage buffer 0
    _Float16* s = Sh[0];
    *(v8h*)(s + (      lr) * 40 + ch) = pre[0]; *(v8h*)(s + (      lr) * 40 + ch + 8) = pre[1];
    *(v8h*)(s + (128 + lr) * 40 + ch) = pre[2]; *(v8h*)(s + (128 + lr) * 40 + ch + 8) = pre[3];
    *(v8h*)(s + (256 + lr) * 40 + ch) = pre[4]; *(v8h*)(s + (256 + lr) * 40 + ch + 8) = pre[5];
  }
  __syncthreads();

  const int NK = K >> 5;
  for (int kt = 0; kt < NK; ++kt) {
    if (kt + 1 < NK) { // prefetch next k-tile while computing this one
      const int k0 = (kt + 1) << 5;
      const float4* a  = (const float4*)(gA  + k0);
      const float4* b0 = (const float4*)(gB0 + k0);
      const float4* b1 = (const float4*)(gB1 + k0);
      pre[0] = cvt8(a[0],  a[1]);  pre[1] = cvt8(a[2],  a[3]);
      pre[2] = cvt8(b0[0], b0[1]); pre[3] = cvt8(b0[2], b0[3]);
      pre[4] = cvt8(b1[0], b1[1]); pre[5] = cvt8(b1[2], b1[3]);
    }

    const _Float16* Asb = Sh[kt & 1];
    const _Float16* Bsb = Asb + 128 * 40;
    v16h af[4], bf[4];
#pragma unroll
    for (int mi = 0; mi < 4; ++mi) af[mi] = ld_frag(Asb + (wm * 64 + mi * 16) * 40, 40, lane);
#pragma unroll
    for (int ni = 0; ni < 4; ++ni) bf[ni] = ld_frag(Bsb + (wn * 64 + ni * 16) * 40, 40, lane);
#pragma unroll
    for (int mi = 0; mi < 4; ++mi)
#pragma unroll
      for (int ni = 0; ni < 4; ++ni)
        acc[mi][ni] = wmma_f16(af[mi], bf[ni], acc[mi][ni]);

    if (kt + 1 < NK) {
      _Float16* s = Sh[(kt + 1) & 1];
      *(v8h*)(s + (      lr) * 40 + ch) = pre[0]; *(v8h*)(s + (      lr) * 40 + ch + 8) = pre[1];
      *(v8h*)(s + (128 + lr) * 40 + ch) = pre[2]; *(v8h*)(s + (128 + lr) * 40 + ch + 8) = pre[3];
      *(v8h*)(s + (256 + lr) * 40 + ch) = pre[4]; *(v8h*)(s + (256 + lr) * 40 + ch + 8) = pre[5];
    }
    __syncthreads();
  }

#pragma unroll
  for (int mi = 0; mi < 4; ++mi) {
    const int row0 = bm + wm * 64 + mi * 16 + ((lane >> 4) << 3);
#pragma unroll
    for (int ni = 0; ni < 4; ++ni) {
      const int col = bn + wn * 64 + ni * 16 + (lane & 15);
      const float bc = bias[col];
#pragma unroll
      for (int r2 = 0; r2 < 8; ++r2) {
        float v = acc[mi][ni][r2] + bc;
        if (ACT == 1) v = 0.5f * v * (1.0f + erff(v * 0.70710678118654752f));
        const size_t idx = (size_t)(row0 + r2) * N + col;
        if (RES) v += res[idx];
        out[idx] = v;
      }
    }
  }
}

// ---------------------------------------------------------------------------
// Fused flash attention per (batch, head, 128-query tile).
// Q/K/V read straight from (MROWS, C) buffers at column offset head*64.
// S = Q K^T via WMMA -> LDS -> online softmax -> P(f16) -> O += P V via WMMA.
// All fragments of each WMMA burst are preloaded into distinct registers so
// the ds_load pipeline runs ahead of the matrix ops (no dscnt-0 stalls).
// ---------------------------------------------------------------------------
#define ATTN_SMEM 158208

__global__ __launch_bounds__(256) void attention_kernel(
    const float* __restrict__ Qg, const float* __restrict__ Kg,
    const float* __restrict__ Vg, float* __restrict__ Og)
{
  extern __shared__ char smem[];
  _Float16* Qs = (_Float16*)smem;          // 128 x 72 halves
  _Float16* Ks = Qs + 128 * 72;            // 128 x 72
  _Float16* Vt = Ks + 128 * 72;            // 64 x 136 (V transposed: [d][n])
  _Float16* Ps = Vt + 64 * 136;            // 128 x 136
  float*    Ss   = (float*)(Ps + 128 * 136); // 128 x 132
  float*    mrow = Ss + 128 * 132;
  float*    lrow = mrow + 128;
  float*    rsc  = lrow + 128;

  const int qt = blockIdx.x, hh = blockIdx.y, bb = blockIdx.z;
  const int tid = threadIdx.x, lane = tid & 31, wid = tid >> 5;
  const int wm = wid >> 2, wn = wid & 3;
  const int r  = tid >> 1, ch = (tid & 1) * 32;

  { // stage Q tile (128 x 64) as f16
    const float4* src = (const float4*)(Qg + (size_t)(bb * NTOK + qt * 128 + r) * CDIM + hh * HD + ch);
    _Float16* dst = Qs + r * 72 + ch;
#pragma unroll
    for (int j = 0; j < 4; ++j) cvt_store8(dst + 8 * j, src[2 * j], src[2 * j + 1]);
  }
  if (tid < 128) { mrow[tid] = -1e30f; lrow[tid] = 0.0f; }

  v8f oacc[4] = {};

  for (int kt = 0; kt < 4; ++kt) {
    __syncthreads();                       // Q/stats ready, prev tiles consumed
    { // stage K tile
      const float4* src = (const float4*)(Kg + (size_t)(bb * NTOK + kt * 128 + r) * CDIM + hh * HD + ch);
      _Float16* dst = Ks + r * 72 + ch;
#pragma unroll
      for (int j = 0; j < 4; ++j) cvt_store8(dst + 8 * j, src[2 * j], src[2 * j + 1]);
    }
    { // stage V tile transposed
      const float4* src = (const float4*)(Vg + (size_t)(bb * NTOK + kt * 128 + r) * CDIM + hh * HD + ch);
#pragma unroll
      for (int j = 0; j < 8; ++j) {
        float4 f = src[j];
        Vt[(ch + 4 * j + 0) * 136 + r] = (_Float16)f.x;
        Vt[(ch + 4 * j + 1) * 136 + r] = (_Float16)f.y;
        Vt[(ch + 4 * j + 2) * 136 + r] = (_Float16)f.z;
        Vt[(ch + 4 * j + 3) * 136 + r] = (_Float16)f.w;
      }
    }
    __syncthreads();

    // S = Q K^T  (wave tile 64x32, K = 64): preload all 12 fragments, then MMA
    v8f sacc[4][2] = {};
    {
      v16h qf[2][4], kf[2][2];
#pragma unroll
      for (int kk2 = 0; kk2 < 2; ++kk2) {
        const int kk = kk2 * 32;
#pragma unroll
        for (int mi = 0; mi < 4; ++mi)
          qf[kk2][mi] = ld_frag(Qs + (wm * 64 + mi * 16) * 72 + kk, 72, lane);
        kf[kk2][0] = ld_frag(Ks + (wn * 32 + 0 ) * 72 + kk, 72, lane);
        kf[kk2][1] = ld_frag(Ks + (wn * 32 + 16) * 72 + kk, 72, lane);
      }
#pragma unroll
      for (int kk2 = 0; kk2 < 2; ++kk2)
#pragma unroll
        for (int mi = 0; mi < 4; ++mi) {
          sacc[mi][0] = wmma_f16(qf[kk2][mi], kf[kk2][0], sacc[mi][0]);
          sacc[mi][1] = wmma_f16(qf[kk2][mi], kf[kk2][1], sacc[mi][1]);
        }
    }
#pragma unroll
    for (int mi = 0; mi < 4; ++mi) {
      const int rr = wm * 64 + mi * 16 + ((lane >> 4) << 3);
#pragma unroll
      for (int ni = 0; ni < 2; ++ni) {
        const int cc = wn * 32 + ni * 16 + (lane & 15);
#pragma unroll
        for (int r2 = 0; r2 < 8; ++r2)
          Ss[(rr + r2) * 132 + cc] = sacc[mi][ni][r2] * 0.125f;  // * HD^-0.5
      }
    }
    __syncthreads();

    // online softmax, one row per thread
    if (tid < 128) {
      const float* srow = Ss + tid * 132;
      const float mold = mrow[tid];
      float mx = mold;
      for (int j = 0; j < 128; ++j) mx = fmaxf(mx, srow[j]);
      const float sc = expf(mold - mx);
      float l = lrow[tid] * sc;
      _Float16* prow = Ps + tid * 136;
      for (int j = 0; j < 128; ++j) {
        const float e = expf(srow[j] - mx);
        l += e;
        prow[j] = (_Float16)e;
      }
      mrow[tid] = mx; lrow[tid] = l; rsc[tid] = sc;
    }
    __syncthreads();

    // rescale running O, then O += P V (wave tile 64x16, K = 128):
    // preload all 20 fragments into distinct registers, then 16 chained MMAs
#pragma unroll
    for (int mi = 0; mi < 4; ++mi) {
      const int rr = wm * 64 + mi * 16 + ((lane >> 4) << 3);
#pragma unroll
      for (int r2 = 0; r2 < 8; ++r2) oacc[mi][r2] *= rsc[rr + r2];
    }
    {
      v16h pf[4][4], vf[4];
#pragma unroll
      for (int kk4 = 0; kk4 < 4; ++kk4) {
        const int kk = kk4 * 32;
        vf[kk4] = ld_frag(Vt + (wn * 16) * 136 + kk, 136, lane);
#pragma unroll
        for (int mi = 0; mi < 4; ++mi)
          pf[kk4][mi] = ld_frag(Ps + (wm * 64 + mi * 16) * 136 + kk, 136, lane);
      }
#pragma unroll
      for (int kk4 = 0; kk4 < 4; ++kk4)
#pragma unroll
        for (int mi = 0; mi < 4; ++mi)
          oacc[mi] = wmma_f16(pf[kk4][mi], vf[kk4], oacc[mi]);
    }
  }

#pragma unroll
  for (int mi = 0; mi < 4; ++mi) {
    const int rr = wm * 64 + mi * 16 + ((lane >> 4) << 3);
    const int cc = wn * 16 + (lane & 15);
#pragma unroll
    for (int r2 = 0; r2 < 8; ++r2) {
      const size_t grow = (size_t)(bb * NTOK + qt * 128 + rr + r2);
      Og[grow * CDIM + hh * HD + cc] = oacc[mi][r2] / lrow[rr + r2];
    }
  }
}

// ---------------------------------------------------------------------------
// Patch embed: one block per token; stages the 512-float cube in LDS.
// ---------------------------------------------------------------------------
__global__ __launch_bounds__(256) void patch_embed_kernel(
    const float* __restrict__ x, const float* __restrict__ pw,
    const float* __restrict__ pbias, const float* __restrict__ pos,
    float* __restrict__ h)
{
  __shared__ float xv[512];
  const int g = blockIdx.x;
  const int b = g >> 9, tt = (g >> 6) & 7, s = g & 63;
  const int gh = s >> 3, gw = s & 7;
  const int tid = threadIdx.x;
  for (int p = tid; p < 512; p += 256) {
    const int pb_ = p >> 6, rem = p & 63, p1 = rem >> 3, p2 = rem & 7;
    xv[p] = x[(((size_t)b * 64 + tt * 8 + pb_) * 64 + gh * 8 + p1) * 64 + gw * 8 + p2];
  }
  __syncthreads();
  for (int c = tid; c < CDIM; c += 256) {
    const float* wr = pw + (size_t)c * 512;
    float acc = 0.f;
    for (int p = 0; p < 512; ++p) acc = fmaf(xv[p], wr[p], acc);
    h[(size_t)g * CDIM + c] = acc + pbias[c] + pos[(size_t)(tt * 64 + s) * CDIM + c];
  }
}

// ---------------------------------------------------------------------------
// LayerNorm over last dim (n = 768 or 6144), one row per block.
// ---------------------------------------------------------------------------
__global__ __launch_bounds__(256) void layernorm_kernel(
    const float* __restrict__ in, const float* __restrict__ w,
    const float* __restrict__ b, float* __restrict__ out, int n)
{
  __shared__ float red[256], red2[256];
  __shared__ float s_mean, s_rstd;
  const int row = blockIdx.x, tid = threadIdx.x;
  const float* x = in + (size_t)row * n;
  float s = 0.f, s2 = 0.f;
  for (int j = tid; j < n; j += 256) { const float v = x[j]; s += v; s2 += v * v; }
  red[tid] = s; red2[tid] = s2; __syncthreads();
  for (int st = 128; st > 0; st >>= 1) {
    if (tid < st) { red[tid] += red[tid + st]; red2[tid] += red2[tid + st]; }
    __syncthreads();
  }
  if (tid == 0) {
    const float m = red[0] / n;
    const float v = red2[0] / n - m * m;
    s_mean = m; s_rstd = rsqrtf(v + 1e-5f);
  }
  __syncthreads();
  const float m = s_mean, rs = s_rstd;
  float* y = out + (size_t)row * n;
  for (int j = tid; j < n; j += 256) y[j] = (x[j] - m) * rs * w[j] + b[j];
}

// feat[(b*64+s)*6144 + t*768 + c] = h[(b*512 + t*64 + s)*768 + c]
__global__ void gather_feat_kernel(const float* __restrict__ h, float* __restrict__ feat) {
  const int i = blockIdx.x * 256 + threadIdx.x;     // i < 512*6144
  const int c = i % CDIM;
  const int t = (i / CDIM) % TB;
  const int s = (i / (CDIM * TB)) % 64;
  const int b = i / (CDIM * TB * 64);
  feat[i] = h[((size_t)b * NTOK + t * 64 + s) * CDIM + c];
}

__global__ void pool_kernel(const float* __restrict__ fln, float* __restrict__ pooled) {
  const int i = blockIdx.x * 256 + threadIdx.x;     // i < 8*6144
  if (i >= BATCH * TB * CDIM) return;
  const int b = i / (TB * CDIM), j = i % (TB * CDIM);
  float s = 0.f;
  for (int ss = 0; ss < 64; ++ss) s += fln[((size_t)b * 64 + ss) * (TB * CDIM) + j];
  pooled[i] = s * (1.0f / 64.0f);
}

__global__ __launch_bounds__(256) void head_kernel(
    const float* __restrict__ pooled, const float* __restrict__ hw,
    const float* __restrict__ hb, float* __restrict__ out)
{
  __shared__ float red[256];
  const int cls = blockIdx.x, b = blockIdx.y, tid = threadIdx.x;
  const float* p = pooled + (size_t)b * (TB * CDIM);
  const float* w = hw + (size_t)cls * (TB * CDIM);
  float s = 0.f;
  for (int j = tid; j < TB * CDIM; j += 256) s = fmaf(p[j], w[j], s);
  red[tid] = s; __syncthreads();
  for (int st = 128; st > 0; st >>= 1) { if (tid < st) red[tid] += red[tid + st]; __syncthreads(); }
  if (tid == 0) out[(size_t)b * 100 + cls] = red[0] + hb[cls];
}

// ---------------------------------------------------------------------------
extern "C" void kernel_launch(void* const* d_in, const int* in_sizes, int n_in,
                              void* d_out, int out_size, void* d_ws, size_t ws_size,
                              hipStream_t stream) {
  (void)in_sizes; (void)n_in; (void)out_size; (void)ws_size;

  const float* x       = (const float*)d_in[0];
  const float* patch_w = (const float*)d_in[1];
  const float* patch_b = (const float*)d_in[2];
  const float* pos_emb = (const float*)d_in[3];
  const float* ln1_w   = (const float*)d_in[4];
  const float* ln1_b   = (const float*)d_in[5];
  const float* qw      = (const float*)d_in[6];
  const float* qbias   = (const float*)d_in[7];
  const float* kw      = (const float*)d_in[8];
  const float* kbias   = (const float*)d_in[9];
  const float* vw      = (const float*)d_in[10];
  const float* vbias   = (const float*)d_in[11];
  const float* pw      = (const float*)d_in[12];
  const float* pbias   = (const float*)d_in[13];
  const float* ln2_w   = (const float*)d_in[14];
  const float* ln2_b   = (const float*)d_in[15];
  const float* f1w     = (const float*)d_in[16];
  const float* f1b     = (const float*)d_in[17];
  const float* f2w     = (const float*)d_in[18];
  const float* f2b     = (const float*)d_in[19];
  const float* fcn_w   = (const float*)d_in[20];
  const float* fcn_b   = (const float*)d_in[21];
  const float* head_w  = (const float*)d_in[22];
  const float* head_b  = (const float*)d_in[23];
  float* out = (float*)d_out;

  const size_t TOK = (size_t)MROWS * CDIM;          // 3,145,728 floats
  float* h    = (float*)d_ws;
  float* abuf = h    + TOK;
  float* qbuf = abuf + TOK;
  float* kbuf = qbuf + TOK;
  float* vbuf = kbuf + TOK;
  float* obuf = vbuf + TOK;
  float* mbuf = qbuf;            // FFN hidden (4 x TOK) aliases q/k/v/o region
  float* feat = qbuf;            // final-stage buffers also reuse that region
  float* fln  = qbuf + TOK;
  float* pooled = qbuf + 2 * TOK;

  (void)hipFuncSetAttribute((const void*)attention_kernel,
                            hipFuncAttributeMaxDynamicSharedMemorySize, ATTN_SMEM);

  patch_embed_kernel<<<MROWS, 256, 0, stream>>>(x, patch_w, patch_b, pos_emb, h);

  const dim3 gC(CDIM / 256, MROWS / 128);   // (3, 32)
  const dim3 gF(FF / 256,   MROWS / 128);   // (12, 32)

  for (int i = 0; i < DEPTH; ++i) {
    layernorm_kernel<<<MROWS, 256, 0, stream>>>(h, ln1_w + i * CDIM, ln1_b + i * CDIM, abuf, CDIM);
    gemm_kernel<0, 0><<<gC, 256, 0, stream>>>(abuf, qw + (size_t)i * CDIM * CDIM,
                                              qbias + i * CDIM, nullptr, qbuf, MROWS, CDIM, CDIM);
    gemm_kernel<0, 0><<<gC, 256, 0, stream>>>(abuf, kw + (size_t)i * CDIM * CDIM,
                                              kbias + i * CDIM, nullptr, kbuf, MROWS, CDIM, CDIM);
    gemm_kernel<0, 0><<<gC, 256, 0, stream>>>(abuf, vw + (size_t)i * CDIM * CDIM,
                                              vbias + i * CDIM, nullptr, vbuf, MROWS, CDIM, CDIM);
    attention_kernel<<<dim3(NTOK / 128, NHEAD, BATCH), 256, ATTN_SMEM, stream>>>(qbuf, kbuf, vbuf, obuf);
    gemm_kernel<0, 1><<<gC, 256, 0, stream>>>(obuf, pw + (size_t)i * CDIM * CDIM,
                                              pbias + i * CDIM, h, h, MROWS, CDIM, CDIM);
    layernorm_kernel<<<MROWS, 256, 0, stream>>>(h, ln2_w + i * CDIM, ln2_b + i * CDIM, abuf, CDIM);
    gemm_kernel<1, 0><<<gF, 256, 0, stream>>>(abuf, f1w + (size_t)i * FF * CDIM,
                                              f1b + i * FF, nullptr, mbuf, MROWS, FF, CDIM);
    gemm_kernel<0, 1><<<gC, 256, 0, stream>>>(mbuf, f2w + (size_t)i * CDIM * FF,
                                              f2b + i * CDIM, h, h, MROWS, CDIM, FF);
  }

  gather_feat_kernel<<<(MROWS * CDIM) / 256, 256, 0, stream>>>(h, feat);
  layernorm_kernel<<<BATCH * 64, 256, 0, stream>>>(feat, fcn_w, fcn_b, fln, TB * CDIM);
  pool_kernel<<<(BATCH * TB * CDIM + 255) / 256, 256, 0, stream>>>(fln, pooled);
  head_kernel<<<dim3(100, BATCH), 256, 0, stream>>>(pooled, head_w, head_b, out);
}
